// LocalSelfAttention_8615704396234
// MI455X (gfx1250) — matrix operand order repaired and checked by reading, not compiled, not run on hardware
//
#include <hip/hip_runtime.h>
#include <hip/hip_bf16.h>

typedef __attribute__((ext_vector_type(16))) __bf16 v16bf;
typedef __attribute__((ext_vector_type(8)))  __bf16 v8bf;
typedef __attribute__((ext_vector_type(8)))  float  v8f;
typedef __attribute__((ext_vector_type(4)))  float  v4f;

#define HWSZ 9216   // 96*96
#define CCH  256
#define OC3  768
#define HDIM 32

// ---------------- helper: WMMA fragment loads (bf16, 16x16x32) ----------------
// A-matrix 16x32 layout: lane (m = l&15, kh = l>>4) holds
// K = kh*16 + 0..15 (matches B fragment order).
__device__ __forceinline__ v16bf load_a_frag(const __bf16* row_kb, int kh) {
    const v8bf lo = *(const v8bf*)(row_kb + kh * 16);
    const v8bf hi = *(const v8bf*)(row_kb + kh * 16 + 8);
    v16bf a;
#pragma unroll
    for (int i = 0; i < 8; ++i) { a[i] = lo[i]; a[i + 8] = hi[i]; }
    return a;
}
// B-matrix 32x16 layout: lane (n = l&15, kh = l>>4) holds K = kh*16 + 0..15
// packed 2/VGPR -> one contiguous 32B run of the weight row.
__device__ __forceinline__ v16bf load_b_frag(const __bf16* row_kb_kh) {
    const v8bf lo = *(const v8bf*)(row_kb_kh);
    const v8bf hi = *(const v8bf*)(row_kb_kh + 8);
    v16bf b;
#pragma unroll
    for (int i = 0; i < 8; ++i) { b[i] = lo[i]; b[i + 8] = hi[i]; }
    return b;
}

// ---------------- prep kernels ----------------
__global__ void cast_f32_bf16(const float* __restrict__ src, __bf16* __restrict__ dst, int n) {
    int i = blockIdx.x * blockDim.x + threadIdx.x;
    if (i < n) dst[i] = (__bf16)src[i];
}

// x [B][C][HW] f32 -> xt [B][HW][C] bf16 (tiled LDS transpose)
__global__ void __launch_bounds__(256) transpose_cast(const float* __restrict__ x,
                                                      __bf16* __restrict__ xt) {
    __shared__ float t[32][33];
    const int b = blockIdx.z;
    const int p0 = blockIdx.x * 32, c0 = blockIdx.y * 32;
    const float* xb = x + (size_t)b * CCH * HWSZ;
#pragma unroll
    for (int cc = 0; cc < 4; ++cc) {
        int c = c0 + threadIdx.y + cc * 8;
        t[threadIdx.x][threadIdx.y + cc * 8] = xb[(size_t)c * HWSZ + p0 + threadIdx.x];
    }
    __syncthreads();
    __bf16* xtb = xt + (size_t)b * HWSZ * CCH;
#pragma unroll
    for (int pp = 0; pp < 4; ++pp) {
        int pl = threadIdx.y + pp * 8;
        xtb[(size_t)(p0 + pl) * CCH + c0 + threadIdx.x] = (__bf16)t[pl][threadIdx.x];
    }
}

// ---------------- GEMM 1: qkv[p,o] = sum_c xt[p,c] * wqkv[o,c]  (f32 out) ----
// block: 8 waves = 2 p-waves x 4 o-waves -> 32p x 256o tile; wave tile 16p x 64o
__global__ void __launch_bounds__(256) gemm_qkv(const __bf16* __restrict__ xt,
                                                const __bf16* __restrict__ wq,
                                                float* __restrict__ qkv) {
    const int b = blockIdx.z;
    const int tid = threadIdx.x, wave = tid >> 5, lane = tid & 31;
    const int pw = wave >> 2, ow = wave & 3;
    const int ptile = blockIdx.x * 32 + pw * 16;
    const int otile = blockIdx.y * 256 + ow * 64;
    const int mn = lane & 15, kh = lane >> 4;
    const __bf16* xrow = xt + (size_t)b * HWSZ * CCH + (size_t)(ptile + mn) * CCH;
    v8f acc[4] = {};
    for (int kb = 0; kb < CCH; kb += 32) {
        if (kb + 32 < CCH) __builtin_prefetch(xrow + kb + 32, 0, 0);
        v16bf a = load_a_frag(xrow + kb, kh);
#pragma unroll
        for (int t = 0; t < 4; ++t) {
            v16bf bf = load_b_frag(wq + (size_t)(otile + t * 16 + mn) * CCH + kb + kh * 16);
            acc[t] = __builtin_amdgcn_wmma_f32_16x16x32_bf16(
                false, a, false, bf, (short)0, acc[t], false, false);
        }
    }
    // C layout: VGPR r -> M = 8*kh + r, N = lane&15
    float* outp = qkv + (size_t)b * HWSZ * OC3;
#pragma unroll
    for (int t = 0; t < 4; ++t)
#pragma unroll
        for (int r = 0; r < 8; ++r)
            outp[(size_t)(ptile + 8 * kh + r) * OC3 + otile + t * 16 + mn] = acc[t][r];
}

// ---------------- attention core (VALU + LDS) ----------------
// wave = 4 pixels of one head, one per pass; lane = n channel; k/v window staged in LDS.
__global__ void __launch_bounds__(256) attn_kernel(const float* __restrict__ qkv,
                                                   __bf16* __restrict__ at) {
    __shared__ float ldsK[3 * 34 * 32];
    __shared__ float ldsV[3 * 34 * 32];
    const int bh = blockIdx.z, b = bh >> 3, h = bh & 7;
    const int i = blockIdx.y;
    const int j0 = blockIdx.x * 32;
    const float* qb = qkv + (size_t)b * HWSZ * OC3;
    const int tid = threadIdx.x;
    for (int idx = tid; idx < 3 * 34 * 32; idx += 256) {
        int ch = idx & 31;
        int cr = idx >> 5;
        int cc = cr % 34, rr = cr / 34;
        int gi = i - 1 + rr, gj = j0 - 1 + cc;
        float kv = 0.f, vv = 0.f;
        if (gi >= 0 && gi < 96 && gj >= 0 && gj < 96) {
            size_t base = (size_t)(gi * 96 + gj) * OC3 + h * HDIM + ch;
            kv = qb[base + 256];
            vv = qb[base + 512];
        }
        ldsK[idx] = kv;
        ldsV[idx] = vv;
    }
    __syncthreads();
    const int wave = tid >> 5, lane = tid & 31;
    const float scale = 0.17677669529663687f;  // 32^-0.5
    for (int px = 0; px < 4; ++px) {
        const int j = j0 + wave * 4 + px;
        const int jl = j - j0;  // 0..31
        float qs[9];
#pragma unroll
        for (int s = 0; s < 9; ++s) {
            int gi = i + s / 3 - 1, gj = j + s % 3 - 1;
            qs[s] = (gi >= 0 && gi < 96 && gj >= 0 && gj < 96)
                        ? qb[(size_t)(gi * 96 + gj) * OC3 + h * HDIM + lane] * scale
                        : 0.f;
        }
        float dots[32];
#pragma unroll
        for (int m = 0; m < 32; ++m) {
            float acc = 0.f;
#pragma unroll
            for (int s = 0; s < 9; ++s)
                acc = fmaf(qs[s], ldsK[((s / 3) * 34 + jl + s % 3) * 32 + m], acc);
            dots[m] = acc;
        }
        float mx = dots[0];
#pragma unroll
        for (int m = 1; m < 32; ++m) mx = fmaxf(mx, dots[m]);
        float se = 0.f;
#pragma unroll
        for (int m = 0; m < 32; ++m) {
            float e = __expf(dots[m] - mx);
            dots[m] = e;
            se += e;
        }
        float o = 0.f;
#pragma unroll
        for (int m = 0; m < 32; ++m) {
            float vs = 0.f;
#pragma unroll
            for (int s = 0; s < 9; ++s) vs += ldsV[((s / 3) * 34 + jl + s % 3) * 32 + m];
            o = fmaf(dots[m], vs, o);
        }
        o /= se;
        at[(size_t)b * HWSZ * CCH + (size_t)(i * 96 + j) * CCH + h * HDIM + lane] = (__bf16)o;
    }
}

// ---------------- GEMM 2: out[o,p] = sum_c at[p,c]*wout[o,c] + bias[o] + x[o,p]
__global__ void __launch_bounds__(256) gemm_out(const __bf16* __restrict__ at,
                                                const __bf16* __restrict__ wo,
                                                const float* __restrict__ bias,
                                                const float* __restrict__ x,
                                                float* __restrict__ out) {
    const int b = blockIdx.y;
    const int tid = threadIdx.x, wave = tid >> 5, lane = tid & 31;
    const int pw = wave >> 2, ow = wave & 3;
    const int ptile = blockIdx.x * 32 + pw * 16;
    const int otile = ow * 64;
    const int mn = lane & 15, kh = lane >> 4;
    const __bf16* arow = at + (size_t)b * HWSZ * CCH + (size_t)(ptile + mn) * CCH;
    v8f acc[4] = {};
    for (int kb = 0; kb < CCH; kb += 32) {
        if (kb + 32 < CCH) __builtin_prefetch(arow + kb + 32, 0, 0);
        v16bf a = load_a_frag(arow + kb, kh);
#pragma unroll
        for (int t = 0; t < 4; ++t) {
            v16bf bf = load_b_frag(wo + (size_t)(otile + t * 16 + mn) * CCH + kb + kh * 16);
            acc[t] = __builtin_amdgcn_wmma_f32_16x16x32_bf16(
                false, a, false, bf, (short)0, acc[t], false, false);
        }
    }
    const int p0 = ptile + 8 * kh;  // lane's 8 values are consecutive positions
#pragma unroll
    for (int t = 0; t < 4; ++t) {
        const int o = otile + t * 16 + mn;
        const float* xr = x + (size_t)(b * CCH + o) * HWSZ + p0;
        float* orow = out + (size_t)(b * CCH + o) * HWSZ + p0;
        const float bo = bias[o];
        v4f r0 = *(const v4f*)xr;
        v4f r1 = *(const v4f*)(xr + 4);
        v4f w0, w1;
#pragma unroll
        for (int q = 0; q < 4; ++q) {
            w0[q] = acc[t][q] + r0[q] + bo;
            w1[q] = acc[t][4 + q] + r1[q] + bo;
        }
        *(v4f*)orow = w0;
        *(v4f*)(orow + 4) = w1;
    }
}

// ---------------- launch ----------------
extern "C" void kernel_launch(void* const* d_in, const int* in_sizes, int n_in,
                              void* d_out, int out_size, void* d_ws, size_t ws_size,
                              hipStream_t stream) {
    const float* x    = (const float*)d_in[0];
    const float* wqkv = (const float*)d_in[1];
    const float* wout = (const float*)d_in[2];
    const float* bout = (const float*)d_in[3];
    float* out = (float*)d_out;

    char* ws = (char*)d_ws;
    size_t off = 0;
    __bf16* xt  = (__bf16*)(ws + off); off += (size_t)2 * HWSZ * CCH * 2;  // 9.4 MB
    __bf16* wqb = (__bf16*)(ws + off); off += (size_t)OC3 * CCH * 2;       // 0.4 MB
    __bf16* wob = (__bf16*)(ws + off); off += (size_t)CCH * CCH * 2;       // 0.13 MB
    float*  qkv = (float*)(ws + off);  off += (size_t)2 * HWSZ * OC3 * 4;  // 56.6 MB
    __bf16* at  = (__bf16*)(ws + off); off += (size_t)2 * HWSZ * CCH * 2;  // 9.4 MB

    cast_f32_bf16<<<(OC3 * CCH + 255) / 256, 256, 0, stream>>>(wqkv, wqb, OC3 * CCH);
    cast_f32_bf16<<<(CCH * CCH + 255) / 256, 256, 0, stream>>>(wout, wob, CCH * CCH);
    transpose_cast<<<dim3(HWSZ / 32, CCH / 32, 2), dim3(32, 8), 0, stream>>>(x, xt);
    gemm_qkv<<<dim3(HWSZ / 32, OC3 / 256, 2), 256, 0, stream>>>(xt, wqb, qkv);
    attn_kernel<<<dim3(3, 96, 16), 256, 0, stream>>>(qkv, at);
    gemm_out<<<dim3(HWSZ / 32, 2), 256, 0, stream>>>(at, wob, bout, x, out);
}
